// GAT_Net_87110526697563
// MI455X (gfx1250) — matrix operand order repaired and checked by reading, not compiled, not run on hardware
//
#include <hip/hip_runtime.h>

#define N_NODES 100000
#define N_EDGES 1200000
#define E_TOT   (N_EDGES + N_NODES)
#define F_IN    128

typedef _Float16 half_t;
typedef __attribute__((ext_vector_type(16))) _Float16 v16h;
typedef __attribute__((ext_vector_type(8)))  _Float16 v8h;
typedef __attribute__((ext_vector_type(8)))  float    v8f;

union V16 { v16h v; v8h h[2]; };

// order-preserving float -> uint encoding for atomicMax-based segment max
__device__ __forceinline__ unsigned enc_f(float f) {
    unsigned u = __float_as_uint(f);
    return (u & 0x80000000u) ? ~u : (u | 0x80000000u);
}
__device__ __forceinline__ float dec_f(unsigned v) {
    return __uint_as_float((v & 0x80000000u) ? (v & 0x7FFFFFFFu) : ~v);
}
#define ENC_NEG_INF 0x007FFFFFu   // enc(-inf)

// ---------------- elementwise convert (optional ELU) fp32 -> f16 ----------------
__global__ void k_convert(const float* __restrict__ in, half_t* __restrict__ out,
                          int count, int do_elu) {
    int i = blockIdx.x * blockDim.x + threadIdx.x;
    if (i >= count) return;
    float x = in[i];
    if (do_elu) x = x > 0.f ? x : (expf(x) - 1.f);
    out[i] = (half_t)x;
}

// ------------- pack fp32 weights [128 x ncol_in] into WMMA B-fragment layout -----
// linear index t = ((kt*NT + nt)*32 + lane)*16 + j ; lane<16 -> K=kt*32+j, else +16
__global__ void k_pack_w(const float* __restrict__ W, int ncol_in, int NT,
                         half_t* __restrict__ Bpack) {
    int total = 4 * NT * 32 * 16;
    for (int t = blockIdx.x * blockDim.x + threadIdx.x; t < total;
         t += blockDim.x * gridDim.x) {
        int j    = t & 15;
        int lane = (t >> 4) & 31;
        int nt   = (t >> 9) % NT;
        int kt   = (t >> 9) / NT;
        int n    = nt * 16 + (lane & 15);
        int k    = kt * 32 + ((lane & 16) ? 16 : 0) + j;
        float v  = (n < ncol_in) ? W[k * ncol_in + n] : 0.f;
        Bpack[t] = (half_t)v;
    }
}

__global__ void k_pad_bias(const float* __restrict__ b, int n_in,
                           float* __restrict__ bpad, int ldo) {
    int i = threadIdx.x;
    if (i < ldo) bpad[i] = (i < n_in) ? b[i] : 0.f;
}

// ---------------- WMMA GEMM: C[nrows x NT*16] = A16[nrows x 128] @ Wpack ----------
// Each wave computes a 32(M) x NT*16(N) tile: two 16-row accumulators share every
// B fragment. B fragments are staged in LDS once per block (ds_load_b128 path).
// __launch_bounds__(256, 1) raises the VGPR budget so the 2xNT accumulator set
// (~170 live VGPRs) is register-resident — no scratch spills in the WMMA loop.
template <int NT>
__global__ void __launch_bounds__(256, 1)
k_gemm(const half_t* __restrict__ A16, const half_t* __restrict__ Bpack,
       float* __restrict__ C, int nrows, int ldc) {
    __shared__ half_t Bs[4 * NT * 32 * 16];  // 32KB (NT=8) / 12KB (NT=3)

    // block-level stage of packed weights into LDS (16B vectors)
    {
        const uint4* src = (const uint4*)Bpack;
        uint4* dst = (uint4*)Bs;
        const int total16 = (4 * NT * 32 * 16) / 8;   // #16-byte chunks
        for (int i = threadIdx.x; i < total16; i += blockDim.x) dst[i] = src[i];
    }
    __syncthreads();

    int wave = (blockIdx.x * blockDim.x + threadIdx.x) >> 5;
    int lane = threadIdx.x & 31;
    int m0 = wave * 32;
    if (m0 >= nrows) return;

    const v8f zero = {0.f, 0.f, 0.f, 0.f, 0.f, 0.f, 0.f, 0.f};
    v8f acc0[NT], acc1[NT];
#pragma unroll
    for (int t = 0; t < NT; ++t) { acc0[t] = zero; acc1[t] = zero; }

    const half_t* arow0 = A16 + (size_t)(m0 + (lane & 15)) * F_IN;
    const half_t* arow1 = arow0 + 16 * F_IN;
    const half_t* bl = Bs + lane * 16;

#pragma unroll
    for (int kt = 0; kt < 4; ++kt) {
        int kb = kt * 32 + ((lane & 16) ? 16 : 0);
        V16 a0, a1;
        a0.h[0] = *(const v8h*)(arow0 + kb);
        a0.h[1] = *(const v8h*)(arow0 + kb + 8);
        a1.h[0] = *(const v8h*)(arow1 + kb);
        a1.h[1] = *(const v8h*)(arow1 + kb + 8);
        if (kt < 3) __builtin_prefetch(arow0 + kb + 32, 0, 3);
#pragma unroll
        for (int nt = 0; nt < NT; ++nt) {
            const v16h b = *(const v16h*)(bl + (size_t)(kt * NT + nt) * 512);
            acc0[nt] = __builtin_amdgcn_wmma_f32_16x16x32_f16(
                false, a0.v, false, b, (short)0, acc0[nt], false, false);
            acc1[nt] = __builtin_amdgcn_wmma_f32_16x16x32_f16(
                false, a1.v, false, b, (short)0, acc1[nt], false, false);
        }
    }
    // C/D layout: VGPR g -> M = g + (lane>=16 ? 8 : 0), N = lane & 15
    int mrow = (lane & 16) ? 8 : 0;
    int col0 = (lane & 15);
#pragma unroll
    for (int nt = 0; nt < NT; ++nt)
#pragma unroll
        for (int g = 0; g < 8; ++g) {
            C[(size_t)(m0 + mrow + g) * ldc + nt * 16 + col0]      = acc0[nt][g];
            C[(size_t)(m0 + 16 + mrow + g) * ldc + nt * 16 + col0] = acc1[nt][g];
        }
}

// ---------------- per-node attention logits -------------------------------------
__global__ void k_scores(const float* __restrict__ Hf, int ldh,
                         const float* __restrict__ a_src, const float* __restrict__ a_dst,
                         float* __restrict__ ssrc, float* __restrict__ sdst,
                         int heads, int ch) {
    int t = blockIdx.x * blockDim.x + threadIdx.x;
    if (t >= N_NODES * heads) return;
    int n = t / heads, h = t % heads;
    const float* hv = Hf + (size_t)n * ldh + h * ch;
    const float* as = a_src + h * ch;
    const float* ad = a_dst + h * ch;
    float s0 = 0.f, s1 = 0.f;
    for (int c = 0; c < ch; ++c) { float v = hv[c]; s0 += v * as[c]; s1 += v * ad[c]; }
    ssrc[t] = s0; sdst[t] = s1;
}

__global__ void k_init_md(unsigned* __restrict__ emx, float* __restrict__ den, int count) {
    int i = blockIdx.x * blockDim.x + threadIdx.x;
    if (i < count) { emx[i] = ENC_NEG_INF; den[i] = 0.f; }
}

__global__ void k_init_out(float* __restrict__ out, const float* __restrict__ bpad,
                           int ldo, int count) {
    int i = blockIdx.x * blockDim.x + threadIdx.x;
    if (i < count) out[i] = bpad[i % ldo];
}

__device__ __forceinline__ void edge_sd(const int* ei, int e, int& s, int& d) {
    if (e < N_EDGES) { s = ei[e]; d = ei[N_EDGES + e]; }
    else             { s = d = e - N_EDGES; }   // appended self-loops
}

// ---------------- segment softmax pass 1: max -----------------------------------
__global__ void k_edge_max(const int* __restrict__ ei, const float* __restrict__ ssrc,
                           const float* __restrict__ sdst, unsigned* __restrict__ emx,
                           int heads) {
    int e = blockIdx.x * blockDim.x + threadIdx.x;
    if (e >= E_TOT) return;
    int s, d; edge_sd(ei, e, s, d);
    for (int h = 0; h < heads; ++h) {
        float lg = ssrc[s * heads + h] + sdst[d * heads + h];
        lg = lg > 0.f ? lg : 0.2f * lg;           // leaky relu
        atomicMax(&emx[d * heads + h], enc_f(lg));
    }
}

// -------- segment softmax pass 2: numerator (stored) + denominator atomics ------
__global__ void k_edge_den(const int* __restrict__ ei, const float* __restrict__ ssrc,
                           const float* __restrict__ sdst, const unsigned* __restrict__ emx,
                           float* __restrict__ den, float* __restrict__ exbuf,
                           int heads) {
    int e = blockIdx.x * blockDim.x + threadIdx.x;
    if (e >= E_TOT) return;
    int s, d; edge_sd(ei, e, s, d);
    for (int h = 0; h < heads; ++h) {
        float lg = ssrc[s * heads + h] + sdst[d * heads + h];
        lg = lg > 0.f ? lg : 0.2f * lg;
        float m = dec_f(emx[d * heads + h]);
        float ex = expf(lg - m);
        exbuf[(size_t)e * heads + h] = ex;
        atomicAdd(&den[d * heads + h], ex);
    }
}

// ---------------- pass 3: alpha * h[src] scatter-add (4 channels / thread) -------
__global__ void k_aggregate(const int* __restrict__ ei, const float* __restrict__ Hf, int ldh,
                            const float* __restrict__ exbuf, const float* __restrict__ den,
                            float* __restrict__ out, int ldo, int heads, int ch, int groups) {
    long long t = (long long)blockIdx.x * blockDim.x + threadIdx.x;
    int e = (int)(t / groups);
    if (e >= E_TOT) return;
    int g = (int)(t % groups);
    int s, d; edge_sd(ei, e, s, d);
    int c0 = g * 4;
    int h = c0 / ch;
    float ex = exbuf[(size_t)e * heads + h];
    float alpha = ex / (den[d * heads + h] + 1e-16f);
    float4 hv = *(const float4*)(Hf + (size_t)s * ldh + c0);
    float* o = out + (size_t)d * ldo + c0;
    atomicAdd(o + 0, alpha * hv.x);
    atomicAdd(o + 1, alpha * hv.y);
    atomicAdd(o + 2, alpha * hv.z);
    atomicAdd(o + 3, alpha * hv.w);
}

// ---------------- final log_softmax over 40 classes -----------------------------
__global__ void k_logsoftmax(const float* __restrict__ agg, int ld, float* __restrict__ out) {
    int n = blockIdx.x * blockDim.x + threadIdx.x;
    if (n >= N_NODES) return;
    const float* r = agg + (size_t)n * ld;
    float m = r[0];
    for (int j = 1; j < 40; ++j) m = fmaxf(m, r[j]);
    float s = 0.f;
    for (int j = 0; j < 40; ++j) s += expf(r[j] - m);
    float ls = m + logf(s);
    for (int j = 0; j < 40; ++j) out[n * 40 + j] = r[j] - ls;
}

extern "C" void kernel_launch(void* const* d_in, const int* in_sizes, int n_in,
                              void* d_out, int out_size, void* d_ws, size_t ws_size,
                              hipStream_t stream) {
    (void)in_sizes; (void)n_in; (void)out_size; (void)ws_size;
    const float* x   = (const float*)d_in[0];
    const int*   ei  = (const int*)  d_in[1];
    const float* W0  = (const float*)d_in[2];
    const float* as0 = (const float*)d_in[3];
    const float* ad0 = (const float*)d_in[4];
    const float* b0  = (const float*)d_in[5];
    const float* W1  = (const float*)d_in[6];
    const float* as1 = (const float*)d_in[7];
    const float* ad1 = (const float*)d_in[8];
    const float* b1  = (const float*)d_in[9];
    const float* W2  = (const float*)d_in[10];
    const float* as2 = (const float*)d_in[11];
    const float* ad2 = (const float*)d_in[12];
    const float* b2  = (const float*)d_in[13];

    char* ws = (char*)d_ws;
    size_t off = 0;
    auto carve = [&](size_t bytes) -> void* {
        void* p = ws + off;
        off = (off + bytes + 255) & ~(size_t)255;
        return p;
    };
    half_t*   A16  = (half_t*)  carve((size_t)N_NODES * F_IN * 2);  // f16 GEMM input
    float*    Hf   = (float*)   carve((size_t)N_NODES * F_IN * 4);  // GEMM output h
    float*    Agg  = (float*)   carve((size_t)N_NODES * F_IN * 4);  // aggregated output
    float*    ssrc = (float*)   carve((size_t)N_NODES * 4 * 4);
    float*    sdst = (float*)   carve((size_t)N_NODES * 4 * 4);
    unsigned* emx  = (unsigned*)carve((size_t)N_NODES * 4 * 4);
    float*    den  = (float*)   carve((size_t)N_NODES * 4 * 4);
    float*    exb  = (float*)   carve((size_t)E_TOT * 4 * 4);       // softmax numerators
    half_t*   Bp   = (half_t*)  carve((size_t)4 * 8 * 32 * 16 * 2);
    float*    bpad = (float*)   carve(128 * 4);

    const int GEMM_BLOCKS = (N_NODES / 32 + 7) / 8;  // 8 waves/block, 32 rows/wave

    auto run_layer = [&](const float* W, int ncol_in, int NT,
                         const float* as, const float* ad, const float* b,
                         int heads, int ch) {
        int ldp = NT * 16;  // padded column count (128 or 48)
        k_pack_w<<<8, 256, 0, stream>>>(W, ncol_in, NT, Bp);
        k_pad_bias<<<1, 64, 0, stream>>>(b, ncol_in, bpad, ldp);
        if (NT == 8)
            k_gemm<8><<<GEMM_BLOCKS, 256, 0, stream>>>(A16, Bp, Hf, N_NODES, ldp);
        else
            k_gemm<3><<<GEMM_BLOCKS, 256, 0, stream>>>(A16, Bp, Hf, N_NODES, ldp);
        int nh = N_NODES * heads;
        k_scores<<<(nh + 255) / 256, 256, 0, stream>>>(Hf, ldp, as, ad, ssrc, sdst, heads, ch);
        k_init_md<<<(nh + 255) / 256, 256, 0, stream>>>(emx, den, nh);
        k_edge_max<<<(E_TOT + 255) / 256, 256, 0, stream>>>(ei, ssrc, sdst, emx, heads);
        k_edge_den<<<(E_TOT + 255) / 256, 256, 0, stream>>>(ei, ssrc, sdst, emx, den, exb, heads);
        int no = N_NODES * ldp;
        k_init_out<<<(no + 255) / 256, 256, 0, stream>>>(Agg, bpad, ldp, no);
        int groups = (heads * ch) / 4;
        long long tt = (long long)E_TOT * groups;
        k_aggregate<<<(unsigned)((tt + 255) / 256), 256, 0, stream>>>(
            ei, Hf, ldp, exb, den, Agg, ldp, heads, ch, groups);
    };

    const int NE = N_NODES * F_IN;
    // layer 0
    k_convert<<<(NE + 255) / 256, 256, 0, stream>>>(x, A16, NE, 0);
    run_layer(W0, 128, 8, as0, ad0, b0, 4, 32);
    // layer 1
    k_convert<<<(NE + 255) / 256, 256, 0, stream>>>(Agg, A16, NE, 1);
    run_layer(W1, 128, 8, as1, ad1, b1, 4, 32);
    // layer 2 (pad 40 -> 48 cols)
    k_convert<<<(NE + 255) / 256, 256, 0, stream>>>(Agg, A16, NE, 1);
    run_layer(W2, 40, 3, as2, ad2, b2, 1, 40);
    // log_softmax -> d_out [N, 40]
    k_logsoftmax<<<(N_NODES + 255) / 256, 256, 0, stream>>>(Agg, 48, (float*)d_out);
}
